// Attn_87436944212102
// MI455X (gfx1250) — matrix operand order repaired and checked by reading, not compiled, not run on hardware
//
#include <hip/hip_runtime.h>

typedef __bf16 bf16_t;
typedef __attribute__((ext_vector_type(16))) bf16_t v16bf;
typedef __attribute__((ext_vector_type(8)))  bf16_t v8bf;
typedef __attribute__((ext_vector_type(8)))  float  v8f;

#define NB    16
#define CIN   256
#define COUT  128
#define HH    2000
#define WD    9
#define HB    16              // h-rows per block
#define PIX   (HB*WD)         // 144 pixels per block (= 9 WMMA M-tiles)
#define HWB   (HH*WD)         // 18000
#define FSTR  132             // feat LDS row stride (floats), padded
#define ASTR  40              // A-stage LDS row stride (bf16), padded

// LDS layout (bytes)
#define FEAT_OFF   0
#define OUTB_OFF   (PIX*FSTR*4)               // 76032
#define AST_OFF    (OUTB_OFF + COUT*PIX*4)    // 149760
#define PB_OFF     AST_OFF                    // P scratch reuses A-stage region
#define SMEM_BYTES (AST_OFF + PIX*ASTR*2)     // 161280 -> 2 WGs per 320KB WGP

__global__ __launch_bounds__(256)
void attn_fused_kernel(const float* __restrict__ X,
                       const float* __restrict__ Wm,
                       const float* __restrict__ bias,
                       float* __restrict__ Out)
{
    extern __shared__ __align__(16) char smem_raw[];
    float*  feat = (float*)(smem_raw + FEAT_OFF);   // [144][FSTR]  post-conv feats
    float*  outb = (float*)(smem_raw + OUTB_OFF);   // [128][144]   final tile
    bf16_t* Ast  = (bf16_t*)(smem_raw + AST_OFF);   // [144][ASTR]  bf16 A staging
    bf16_t* Pb   = (bf16_t*)(smem_raw + PB_OFF);    // [8 waves][16][16] P scratch

    const int tid   = threadIdx.x;
    const int wave  = tid >> 5;
    const int lane  = tid & 31;
    const int l16   = lane & 15;
    const int lhalf = lane >> 4;

    const int blk = blockIdx.x;
    const int b   = blk / (HH/HB);
    const int h0  = (blk % (HH/HB)) * HB;

    const long inBase  = (long)b * CIN  * HWB + (long)h0 * WD;
    const long outBase = (long)b * COUT * HWB + (long)h0 * WD;

    const v8f zero8 = {0.f,0.f,0.f,0.f,0.f,0.f,0.f,0.f};

    // ------------- Stage 1: 1x1 conv as GEMM, bf16 WMMA, fp32 accumulate ----
    v8f acc[9];
    #pragma unroll
    for (int m = 0; m < 9; ++m) acc[m] = zero8;

    const int   chN   = wave*16 + l16;        // output-channel column (N) of lane
    const float biasv = bias[chN];
    const int   koffB = lhalf ? 16 : 0;       // contiguous K order for A and B here: 0-15 | 16-31
    const int   koffA = lhalf ? 8  : 0;       // 16-bit A layout: K 0-7/16-23 | 8-15/24-31

    for (int ks = 0; ks < 8; ++ks) {
        const int k0 = ks * 32;

        // stage 32c x 144p slab of X into LDS (bf16, transposed to [p][c]);
        // 1152 float4 = 32c * 36 per block, b128 global loads
        #pragma unroll
        for (int i = 0; i < 5; ++i) {
            int idx4 = tid + 256*i;
            if (i == 4 && tid >= 128) break;          // 1152 = 4*256 + 128
            int c  = idx4 / 36;
            int q  = idx4 - c*36;
            int p0 = q*4;
            float4 x4 = *(const float4*)(X + inBase + (long)(k0 + c)*HWB + p0);
            bf16_t* as = Ast + p0*ASTR + c;
            as[0*ASTR] = (bf16_t)x4.x;
            as[1*ASTR] = (bf16_t)x4.y;
            as[2*ASTR] = (bf16_t)x4.z;
            as[3*ASTR] = (bf16_t)x4.w;
        }

        // speculative prefetch of next k-slab (288 cache lines; counter-free)
        if (ks < 7) {
            int c1 = tid / 9, s1 = tid - c1*9;
            __builtin_prefetch(X + inBase + (long)(k0 + 32 + c1)*HWB + s1*16, 0, 0);
            int t2 = tid + 256;
            int c2 = t2 / 9, s2 = t2 - c2*9;
            if (c2 < 32)
                __builtin_prefetch(X + inBase + (long)(k0 + 32 + c2)*HWB + s2*16, 0, 0);
        }
        __syncthreads();

        // B fragment: this lane's column of W^T (16 contiguous floats from L2)
        v16bf bfrag;
        {
            const float4* wp4 = (const float4*)(Wm + chN*CIN + k0 + koffB);
            float wv[16];
            *(float4*)(&wv[0])  = wp4[0];
            *(float4*)(&wv[4])  = wp4[1];
            *(float4*)(&wv[8])  = wp4[2];
            *(float4*)(&wv[12]) = wp4[3];
            #pragma unroll
            for (int i = 0; i < 16; ++i) bfrag[i] = (bf16_t)wv[i];
        }

        #pragma unroll
        for (int m = 0; m < 9; ++m) {
            const bf16_t* ar = Ast + (m*16 + l16)*ASTR + koffB;
            v8bf lo = *(const v8bf*)(ar);
            v8bf hi = *(const v8bf*)(ar + 8);
            v16bf afrag = __builtin_shufflevector(lo, hi,
                0,1,2,3,4,5,6,7,8,9,10,11,12,13,14,15);
            acc[m] = __builtin_amdgcn_wmma_f32_16x16x32_bf16(
                false, afrag, false, bfrag, (short)0, acc[m], false, false);
        }
        __syncthreads();
    }

    // epilogue: bias add, scatter C-layout -> feat[pixel][ch]
    #pragma unroll
    for (int m = 0; m < 9; ++m) {
        #pragma unroll
        for (int r = 0; r < 8; ++r) {
            int prow = m*16 + r + lhalf*8;
            feat[prow*FSTR + chN] = acc[m][r] + biasv;
        }
    }
    __syncthreads();

    // ------------- Stage 2: grouped attention over the 9 w-positions --------
    const float scale = 0.17677669529663687f;   // 1/sqrt(32)
    bf16_t* pb = Pb + wave*256;                 // per-wave private scratch

    for (int it = 0; it < 8; ++it) {
        const int prob = wave + it*8;           // 64 problems: 16 rows x 4 groups
        const int r0   = prob >> 2;
        const int gi   = prob & 3;
        const float* g = feat + (r0*WD)*FSTR + gi*32;   // g[w*FSTR + d]

        // scores = g @ g^T  (one 16x16x32 bf16 WMMA; rows/cols 9..15 zeroed)
        v16bf sa, sb;
        if (l16 < 9) {
            const float* gr = g + l16*FSTR + koffB;     // A: row w = l16
            #pragma unroll
            for (int i = 0; i < 8; ++i) { sa[i] = (bf16_t)gr[i]; sa[8+i] = (bf16_t)gr[8+i]; }
            const float* gc = g + l16*FSTR + koffB;     // B: col v = l16 (row of g)
            #pragma unroll
            for (int i = 0; i < 16; ++i) sb[i] = (bf16_t)gc[i];
        } else {
            #pragma unroll
            for (int i = 0; i < 16; ++i) { sa[i] = (bf16_t)0.f; sb[i] = (bf16_t)0.f; }
        }
        v8f sc = __builtin_amdgcn_wmma_f32_16x16x32_bf16(
            false, sa, false, sb, (short)0, zero8, false, false);

        // masked softmax over v; v == lane index within each 16-lane half
        #pragma unroll
        for (int r = 0; r < 8; ++r) {
            float s  = (l16 < 9) ? sc[r]*scale : -__builtin_inff();
            float mx = s;
            mx = fmaxf(mx, __shfl_xor(mx, 1, 16));
            mx = fmaxf(mx, __shfl_xor(mx, 2, 16));
            mx = fmaxf(mx, __shfl_xor(mx, 4, 16));
            mx = fmaxf(mx, __shfl_xor(mx, 8, 16));
            float e = (l16 < 9) ? __expf(s - mx) : 0.f;
            float sum = e;
            sum += __shfl_xor(sum, 1, 16);
            sum += __shfl_xor(sum, 2, 16);
            sum += __shfl_xor(sum, 4, 16);
            sum += __shfl_xor(sum, 8, 16);
            sc[r] = e / sum;
        }

        // reshape P: C-layout -> A-layout via per-wave LDS (same-wave DS ops in order)
        #pragma unroll
        for (int r = 0; r < 8; ++r)
            pb[(r + lhalf*8)*16 + l16] = (bf16_t)sc[r];
        v16bf pa;
        #pragma unroll
        for (int i = 0; i < 16; ++i) pa[i] = lhalf ? (bf16_t)0.f : pb[l16*16 + i];  // K 16..31 = 0

        // out = P @ elu(g): M=16(w) K=16(v, padded to 32) N=2x16(d)
        #pragma unroll
        for (int nh = 0; nh < 2; ++nh) {
            v16bf eb;
            if (lhalf == 0) {                   // lanes 0-15 carry K=0..15 (=v)
                const float* gc = g + nh*16 + l16;      // column d of g
                #pragma unroll
                for (int v2 = 0; v2 < 16; ++v2) {
                    float x = (v2 < 9) ? gc[v2*FSTR] : 0.f;
                    float e = (x > 0.f) ? x : (__expf(x) - 1.f);
                    eb[v2] = (bf16_t)e;
                }
            } else {                            // lanes 16-31 carry K=16..31 -> 0
                #pragma unroll
                for (int i = 0; i < 16; ++i) eb[i] = (bf16_t)0.f;
            }
            v8f o = __builtin_amdgcn_wmma_f32_16x16x32_bf16(
                false, pa, false, eb, (short)0, zero8, false, false);

            const int ch = gi*32 + nh*16 + l16;
            float* ob = outb + ch*PIX + r0*WD;
            if (lhalf == 0) {                   // M = 0..7 -> w = 0..7
                #pragma unroll
                for (int r = 0; r < 8; ++r) ob[r] = o[r];
            } else {                            // M = 8 -> w = 8 (9..15 dropped)
                ob[8] = o[0];
            }
        }
    }
    __syncthreads();

    // ------------- Stage 3: async LDS->global stores (ASYNCcnt tracked) -----
    // outb is laid out [ch][144]: contiguous 16B chunks on both sides.
    {
        const unsigned ldsOut = (unsigned)(size_t)(void*)outb;  // LDS byte address
        #pragma unroll
        for (int i = 0; i < 18; ++i) {
            int idx = tid + 256*i;              // 4608 = 128ch * 36 float4
            int ch  = idx / 36;
            int j   = idx - ch*36;
            unsigned long long ga =
                (unsigned long long)(Out + outBase + (long)ch*HWB + j*4);
            unsigned la = ldsOut + (unsigned)((ch*PIX + j*4) * 4);
            asm volatile("global_store_async_from_lds_b128 %0, %1, off"
                         :: "v"(ga), "v"(la) : "memory");
        }
        asm volatile("s_wait_asynccnt 0x0" ::: "memory");
    }
}

extern "C" void kernel_launch(void* const* d_in, const int* in_sizes, int n_in,
                              void* d_out, int out_size, void* d_ws, size_t ws_size,
                              hipStream_t stream)
{
    (void)in_sizes; (void)n_in; (void)out_size; (void)d_ws; (void)ws_size;
    const float* X  = (const float*)d_in[0];   // [16,256,2000,9] fp32
    const float* Wm = (const float*)d_in[1];   // [128,256] fp32
    const float* bv = (const float*)d_in[2];   // [128] fp32
    float* Out = (float*)d_out;                // [16,128,2000,9] fp32

    const int grid = NB * (HH / HB);           // 16 * 125 = 2000 blocks
    attn_fused_kernel<<<grid, 256, SMEM_BYTES, stream>>>(X, Wm, bv, Out);
}
